// MoeTransformer_79474074845412
// MI455X (gfx1250) — matrix operand order repaired and checked by reading, not compiled, not run on hardware
//
#include <hip/hip_runtime.h>
#include <math.h>

typedef __bf16 bf16_t;
typedef __attribute__((ext_vector_type(16))) __bf16 v16bf;
typedef __attribute__((ext_vector_type(8)))  __bf16 v8bf;
typedef __attribute__((ext_vector_type(4)))  __bf16 v4bf;
typedef __attribute__((ext_vector_type(8)))  float  v8f;

#define S_TOK 4096
#define DM    2048
#define NE    8
#define DF    8192
#define CAP   1024   // ceil(2*S/E * 1.0)

#if __has_builtin(__builtin_amdgcn_s_wait_asynccnt)
#define WAIT_ASYNC() __builtin_amdgcn_s_wait_asynccnt(0)
#else
#define WAIT_ASYNC() asm volatile("s_wait_asynccnt 0" ::: "memory")
#endif

// ---------------------------------------------------------------- gating ----

__global__ void moe_logits(const float* __restrict__ x, const float* __restrict__ wg,
                           float* __restrict__ logits) {
  int s = blockIdx.x;
  int tid = threadIdx.x;
  float acc[NE];
#pragma unroll
  for (int e = 0; e < NE; ++e) acc[e] = 0.f;
  const float* xr = x + (size_t)s * DM;
  for (int m = tid; m < DM; m += 256) {
    float xv = xr[m];
    const float4* w4 = (const float4*)(wg + (size_t)m * NE);
    float4 a = w4[0], b = w4[1];
    acc[0] += xv * a.x; acc[1] += xv * a.y; acc[2] += xv * a.z; acc[3] += xv * a.w;
    acc[4] += xv * b.x; acc[5] += xv * b.y; acc[6] += xv * b.z; acc[7] += xv * b.w;
  }
  __shared__ float sm[256 * NE];
#pragma unroll
  for (int e = 0; e < NE; ++e) sm[tid * NE + e] = acc[e];
  __syncthreads();
  for (int st = 128; st > 0; st >>= 1) {
    if (tid < st)
#pragma unroll
      for (int e = 0; e < NE; ++e) sm[tid * NE + e] += sm[(tid + st) * NE + e];
    __syncthreads();
  }
  if (tid < NE) logits[(size_t)s * NE + tid] = sm[tid];
}

__global__ void moe_top2(const float* __restrict__ logits, int* __restrict__ idx1,
                         int* __restrict__ idx2, float* __restrict__ gate1,
                         float* __restrict__ gate2) {
  int s = blockIdx.x * blockDim.x + threadIdx.x;
  if (s >= S_TOK) return;
  float l[NE];
#pragma unroll
  for (int e = 0; e < NE; ++e) l[e] = logits[(size_t)s * NE + e];
  float mx = l[0];
#pragma unroll
  for (int e = 1; e < NE; ++e) mx = fmaxf(mx, l[e]);
  float ex[NE]; float sum = 0.f;
#pragma unroll
  for (int e = 0; e < NE; ++e) { ex[e] = __expf(l[e] - mx); sum += ex[e]; }
  float inv = 1.0f / sum;
  int i1 = 0; float b1 = l[0];
#pragma unroll
  for (int e = 1; e < NE; ++e) if (l[e] > b1) { b1 = l[e]; i1 = e; }   // first-occurrence argmax
  int i2 = -1; float b2 = -3.0e38f;
#pragma unroll
  for (int e = 0; e < NE; ++e) {
    if (e == i1) continue;
    if (l[e] > b2) { b2 = l[e]; i2 = e; }
  }
  idx1[s] = i1; idx2[s] = i2;
  gate1[s] = ex[i1] * inv;
  gate2[s] = ex[i2] * inv;
}

// sequential capacity scan: matches cumsum(mask) semantics of the reference
__global__ void moe_scan(const int* __restrict__ idx1, const int* __restrict__ idx2,
                         int* __restrict__ pos1, int* __restrict__ pos2) {
  __shared__ int c1[NE];
  int e = threadIdx.x;
  if (e < NE) {
    int cnt = 0;
    for (int s = 0; s < S_TOK; ++s)
      if (idx1[s] == e) { pos1[s] = cnt; ++cnt; }   // pre-drop occurrence index
    c1[e] = cnt;                                    // pre-drop total (sum(mask1))
  }
  __syncthreads();
  if (e < NE) {
    int base = c1[e], cnt = 0;
    for (int s = 0; s < S_TOK; ++s)
      if (idx2[s] == e) { pos2[s] = base + cnt; ++cnt; }
  }
}

__global__ void moe_init_slots(int* __restrict__ slot_token) {
  int i = blockIdx.x * blockDim.x + threadIdx.x;
  if (i < NE * CAP) slot_token[i] = -1;
}

__global__ void moe_finalize(const int* __restrict__ idx1, const int* __restrict__ idx2,
                             const float* __restrict__ gate1, const float* __restrict__ gate2,
                             const int* __restrict__ pos1, const int* __restrict__ pos2,
                             int* __restrict__ slot_token,
                             int* __restrict__ tslot1, int* __restrict__ tslot2,
                             float* __restrict__ tg1, float* __restrict__ tg2) {
  int s = blockIdx.x * blockDim.x + threadIdx.x;
  if (s >= S_TOK) return;
  int p1 = pos1[s], p2 = pos2[s];
  bool k1 = p1 < CAP, k2 = p2 < CAP;
  float g1 = k1 ? gate1[s] : 0.f;
  float g2 = k2 ? gate2[s] : 0.f;
  float d = fmaxf(g1 + g2, 1e-9f);
  g1 /= d; g2 /= d;
  int sl1 = k1 ? idx1[s] * CAP + p1 : -1;
  int sl2 = k2 ? idx2[s] * CAP + p2 : -1;
  if (sl1 >= 0) slot_token[sl1] = s;
  if (sl2 >= 0) slot_token[sl2] = s;
  tslot1[s] = sl1; tslot2[s] = sl2;
  tg1[s] = g1; tg2[s] = g2;
}

// gather tokens into [E*CAP, DM] bf16 (zero rows for empty slots)
__global__ void moe_dispatch(const float* __restrict__ x, const int* __restrict__ slot_token,
                             bf16_t* __restrict__ disp) {
  int slot = blockIdx.x;
  int tid = threadIdx.x;
  int tok = slot_token[slot];
  bf16_t* drow = disp + (size_t)slot * DM;
  v8bf v;
  if (tok < 0) {
#pragma unroll
    for (int t = 0; t < 8; ++t) v[t] = (bf16_t)0.0f;
  } else {
    const float4* xr = (const float4*)(x + (size_t)tok * DM);
    float4 a = xr[tid * 2], b = xr[tid * 2 + 1];
    v[0] = (bf16_t)a.x; v[1] = (bf16_t)a.y; v[2] = (bf16_t)a.z; v[3] = (bf16_t)a.w;
    v[4] = (bf16_t)b.x; v[5] = (bf16_t)b.y; v[6] = (bf16_t)b.z; v[7] = (bf16_t)b.w;
  }
  ((v8bf*)drow)[tid] = v;   // 256 threads * 8 bf16 = 2048
}

// ------------------------------------------------------------- WMMA GEMM ----
// Out[e, c, n] = (gelu?) sum_k A[e, c, k] * B[e, k, n]
// A bf16 [E, CAP, KTOT] staged via async global->LDS DMA (ASYNCcnt);
// B fp32 [E, KTOT, NTOT] converted to bf16 while staging; double-buffered LDS.
// blockIdx.x = c-tile (fastest) so the 8 consumers of each weight panel are
// adjacent in launch order -> each wi/wo panel is fetched from HBM once.
template <int KTOT, int NTOT, bool GELU>
__global__ __launch_bounds__(256)
void moe_gemm_bf16(const bf16_t* __restrict__ A, const float* __restrict__ B,
                   bf16_t* __restrict__ OutBf, float* __restrict__ OutF) {
  constexpr int BM = 128, BN = 128, BK = 32;
  constexpr int LDT = BK + 8;               // padded LDS row stride (elements)
  constexpr int NK = KTOT / BK;
  __shared__ bf16_t As[2][BM * LDT];        // [m][k]
  __shared__ bf16_t Bs[2][BN * LDT];        // transposed: [n][k]

  int e  = blockIdx.z;
  int c0 = blockIdx.x * BM;                 // c-tile varies fastest
  int n0 = blockIdx.y * BN;
  const bf16_t* Ablk = A + ((size_t)e * CAP + c0) * KTOT;
  const float*  Bblk = B + (size_t)e * KTOT * NTOT + n0;

  int tid  = threadIdx.x;
  int lane = tid & 31;
  int wave = tid >> 5;        // 0..7
  int wr   = wave & 3;        // 4 row groups of 32
  int wc   = wave >> 2;       // 2 col groups of 64
  int half = lane >> 4;       // lane half
  int l16  = lane & 15;
  int klo  = half * 8;        // A fragment K interleave per ISA layout

  // A staging: 2 threads per row, 32B per thread, async DMA to LDS
  int arow = tid >> 1;
  int akh  = (tid & 1) * 16;
  const bf16_t* aGsrc = Ablk + (size_t)arow * KTOT + akh;
  unsigned aLds[2];
  aLds[0] = (unsigned)(size_t)&As[0][arow * LDT + akh];
  aLds[1] = (unsigned)(size_t)&As[1][arow * LDT + akh];

  // B staging: 4k x 4n micro-tile per thread (8 k-groups x 32 n-groups = 256)
  int bk0t = (tid & 7) * 4;                 // 0..28
  int bn0t = (tid >> 3) * 4;                // 0..124
  const float* bGsrc = Bblk + (size_t)bk0t * NTOT + bn0t;

  v8f acc[2][4];
#pragma unroll
  for (int i = 0; i < 2; ++i)
#pragma unroll
    for (int j = 0; j < 4; ++j)
#pragma unroll
      for (int r = 0; r < 8; ++r) acc[i][j][r] = 0.0f;

  // ---- preload tile 0 ----
  {
    const void* g = (const void*)aGsrc;
    asm volatile("global_load_async_to_lds_b128 %0, %1, off\n\t"
                 "global_load_async_to_lds_b128 %0, %1, off offset:16"
                 :: "v"(aLds[0]), "v"(g) : "memory");
    float bv[4][4];
#pragma unroll
    for (int dk = 0; dk < 4; ++dk) {
      float4 t = *(const float4*)(bGsrc + (size_t)dk * NTOT);
      bv[dk][0] = t.x; bv[dk][1] = t.y; bv[dk][2] = t.z; bv[dk][3] = t.w;
    }
#pragma unroll
    for (int dn = 0; dn < 4; ++dn) {
      v4bf p;
#pragma unroll
      for (int dk = 0; dk < 4; ++dk) p[dk] = (bf16_t)bv[dk][dn];
      *(v4bf*)&Bs[0][(bn0t + dn) * LDT + bk0t] = p;
    }
  }
  WAIT_ASYNC();
  __syncthreads();

  // ---- main loop: compute buf(kt&1), stage buf((kt+1)&1) ----
  for (int kt = 0; kt < NK; ++kt) {
    int cur = kt & 1;
    int nxt = kt + 1;
    float bv[4][4];
    if (nxt < NK) {
      const void* g = (const void*)(aGsrc + (size_t)nxt * BK);
      asm volatile("global_load_async_to_lds_b128 %0, %1, off\n\t"
                   "global_load_async_to_lds_b128 %0, %1, off offset:16"
                   :: "v"(aLds[nxt & 1]), "v"(g) : "memory");
      const float* bg = bGsrc + (size_t)nxt * BK * NTOT;
#pragma unroll
      for (int dk = 0; dk < 4; ++dk) {
        float4 t = *(const float4*)(bg + (size_t)dk * NTOT);
        bv[dk][0] = t.x; bv[dk][1] = t.y; bv[dk][2] = t.z; bv[dk][3] = t.w;
      }
      if (kt + 2 < NK)  // keep the HBM weight stream in flight
        __builtin_prefetch(bGsrc + (size_t)(kt + 2) * BK * NTOT, 0, 0);
    }

    // fragments (ISA VGPR layouts)
    v16bf afrag[2], bfrag[4];
#pragma unroll
    for (int i = 0; i < 2; ++i) {
      const bf16_t* ap = &As[cur][(wr * 32 + i * 16 + l16) * LDT];
      v8bf lo = *(const v8bf*)(ap + klo);
      v8bf hi = *(const v8bf*)(ap + 16 + klo);
#pragma unroll
      for (int t = 0; t < 8; ++t) { afrag[i][t] = lo[t]; afrag[i][t + 8] = hi[t]; }
    }
    int bk0 = half * 8;
#pragma unroll
    for (int j = 0; j < 4; ++j) {
      const bf16_t* bp = &Bs[cur][(wc * 64 + j * 16 + l16) * LDT + bk0];
      v8bf lo = *(const v8bf*)(bp);
      v8bf hi = *(const v8bf*)(bp + 16);
#pragma unroll
      for (int t = 0; t < 8; ++t) { bfrag[j][t] = lo[t]; bfrag[j][t + 8] = hi[t]; }
    }

#pragma unroll
    for (int i = 0; i < 2; ++i)
#pragma unroll
      for (int j = 0; j < 4; ++j)
        acc[i][j] = __builtin_amdgcn_wmma_f32_16x16x32_bf16(
            false, afrag[i], false, bfrag[j], (short)0, acc[i][j], false, false);

    if (nxt < NK) {
#pragma unroll
      for (int dn = 0; dn < 4; ++dn) {
        v4bf p;
#pragma unroll
        for (int dk = 0; dk < 4; ++dk) p[dk] = (bf16_t)bv[dk][dn];
        *(v4bf*)&Bs[nxt & 1][(bn0t + dn) * LDT + bk0t] = p;
      }
    }
    WAIT_ASYNC();
    __syncthreads();
  }

  // epilogue: D layout -> row m = 8*lanehalf + vgpr, col n = lane&15
#pragma unroll
  for (int i = 0; i < 2; ++i)
#pragma unroll
    for (int j = 0; j < 4; ++j)
#pragma unroll
      for (int r = 0; r < 8; ++r) {
        int m = wr * 32 + i * 16 + half * 8 + r;
        int n = wc * 64 + j * 16 + l16;
        float v = acc[i][j][r];
        size_t off = ((size_t)e * CAP + c0 + m) * NTOT + (n0 + n);
        if (GELU) {
          // 0.5*v*(1+tanh(t)) == v * sigmoid(2t); 2t = 1.59577*v + 0.0713548*v^3
          float u = v * (1.5957691216057308f + 0.0713548162726009f * v * v);
          float g = v * (1.0f / (1.0f + __expf(-u)));
          OutBf[off] = (bf16_t)g;
        } else {
          OutF[off] = v;
        }
      }
}

// ---------------------------------------------------------------- combine ---

__global__ void moe_combine(const float* __restrict__ EO, const int* __restrict__ tslot1,
                            const int* __restrict__ tslot2, const float* __restrict__ tg1,
                            const float* __restrict__ tg2, float* __restrict__ out) {
  int s = blockIdx.x;
  int tid = threadIdx.x;
  int s1 = tslot1[s], s2 = tslot2[s];
  float g1 = tg1[s], g2 = tg2[s];
  float4* o = (float4*)(out + (size_t)s * DM);
  const float4* r1 = (s1 >= 0) ? (const float4*)(EO + (size_t)s1 * DM) : nullptr;
  const float4* r2 = (s2 >= 0) ? (const float4*)(EO + (size_t)s2 * DM) : nullptr;
  for (int q = tid; q < DM / 4; q += 256) {
    float4 v = {0.f, 0.f, 0.f, 0.f};
    if (r1) { float4 a = r1[q]; v.x += g1 * a.x; v.y += g1 * a.y; v.z += g1 * a.z; v.w += g1 * a.w; }
    if (r2) { float4 b = r2[q]; v.x += g2 * b.x; v.y += g2 * b.y; v.z += g2 * b.z; v.w += g2 * b.w; }
    o[q] = v;
  }
}

// ----------------------------------------------------------------- launch ---

extern "C" void kernel_launch(void* const* d_in, const int* in_sizes, int n_in,
                              void* d_out, int out_size, void* d_ws, size_t ws_size,
                              hipStream_t stream) {
  const float* x  = (const float*)d_in[0];
  const float* wg = (const float*)d_in[1];
  const float* wi = (const float*)d_in[2];
  const float* wo = (const float*)d_in[3];
  float* out = (float*)d_out;

  char* ws = (char*)d_ws;
  size_t off = 0;
  auto alloc = [&](size_t bytes) -> char* {
    char* p = ws + off;
    off += (bytes + 255) & ~(size_t)255;
    return p;
  };
  float* logits     = (float*)alloc((size_t)S_TOK * NE * 4);
  int*   idx1       = (int*)alloc((size_t)S_TOK * 4);
  int*   idx2       = (int*)alloc((size_t)S_TOK * 4);
  float* gate1      = (float*)alloc((size_t)S_TOK * 4);
  float* gate2      = (float*)alloc((size_t)S_TOK * 4);
  int*   pos1       = (int*)alloc((size_t)S_TOK * 4);
  int*   pos2       = (int*)alloc((size_t)S_TOK * 4);
  int*   tslot1     = (int*)alloc((size_t)S_TOK * 4);
  int*   tslot2     = (int*)alloc((size_t)S_TOK * 4);
  float* tg1        = (float*)alloc((size_t)S_TOK * 4);
  float* tg2        = (float*)alloc((size_t)S_TOK * 4);
  int*   slot_token = (int*)alloc((size_t)NE * CAP * 4);
  bf16_t* disp      = (bf16_t*)alloc((size_t)NE * CAP * DM * 2);   //  32 MB
  bf16_t* H         = (bf16_t*)alloc((size_t)NE * CAP * DF * 2);   // 128 MB
  float*  EO        = (float*)alloc((size_t)NE * CAP * DM * 4);    //  64 MB

  moe_logits<<<S_TOK, 256, 0, stream>>>(x, wg, logits);
  moe_top2<<<S_TOK / 256, 256, 0, stream>>>(logits, idx1, idx2, gate1, gate2);
  moe_init_slots<<<(NE * CAP + 255) / 256, 256, 0, stream>>>(slot_token);
  moe_scan<<<1, 32, 0, stream>>>(idx1, idx2, pos1, pos2);
  moe_finalize<<<S_TOK / 256, 256, 0, stream>>>(idx1, idx2, gate1, gate2, pos1, pos2,
                                                slot_token, tslot1, tslot2, tg1, tg2);
  moe_dispatch<<<NE * CAP, 256, 0, stream>>>(x, slot_token, disp);

  dim3 g1(CAP / 128, DF / 128, NE);   // c-tile fastest: 8 x 64 x 8
  moe_gemm_bf16<DM, DF, true><<<g1, 256, 0, stream>>>(disp, wi, H, nullptr);
  dim3 g2(CAP / 128, DM / 128, NE);   // c-tile fastest: 8 x 16 x 8
  moe_gemm_bf16<DF, DM, false><<<g2, 256, 0, stream>>>(H, wo, nullptr, EO);

  moe_combine<<<S_TOK, 256, 0, stream>>>(EO, tslot1, tslot2, tg1, tg2, out);
}
